// Attention_1563368096483
// MI455X (gfx1250) — compile-verified, matrix operand order checked by reading, not run on hardware
//
#include <hip/hip_runtime.h>
#include <hip/hip_bf16.h>
#include <cstdint>
#include <cstddef>

// ---------------------------------------------------------------------------
// CDNA5 (gfx1250) talking-heads attention.
// All matmuls on v_wmma_f32_16x16x32_f16; Q/K tiles via pipelined Tensor Data
// Mover; Q staged in LDS (register budget: 12 waves must stay resident).
// ---------------------------------------------------------------------------

typedef __attribute__((ext_vector_type(16))) _Float16 v16h;
typedef __attribute__((ext_vector_type(8)))  _Float16 v8h;
typedef __attribute__((ext_vector_type(8)))  float    v8f;
typedef __attribute__((ext_vector_type(4)))  float    v4f;

union AFrag { v16h v; v8h h[2]; };

__device__ __forceinline__ v8f wmma_f16(const AFrag& a, const AFrag& b, v8f c) {
    // D = A(16x32 f16) x B(32x16 f16) + C(16x16 f32)
    return __builtin_amdgcn_wmma_f32_16x16x32_f16(
        /*neg_a=*/false, a.v, /*neg_b=*/false, b.v,
        /*c_mod=*/(short)0, c, /*reuse_a=*/false, /*reuse_b=*/false);
}

static constexpr int HEADS = 12;
static constexpr int DH    = 64;
static constexpr int NQ    = 1024;
static constexpr int NKV   = 2048;
static constexpr int DM    = 768;      // model dim == inner dim
static constexpr int QT    = 32;       // query tile rows (lane == query row)
static constexpr int JT    = 32;       // key/value tile rows (== WMMA K)
static constexpr float SCALE = 0.125f; // 64^-0.5

// ---------------------------------------------------------------------------
// Tensor Data Mover: 2D f16 tile load, global -> LDS (async, TENSORcnt).
// ---------------------------------------------------------------------------
#if defined(__has_builtin)
#  if __has_builtin(__builtin_amdgcn_tensor_load_to_lds)
#    define HAVE_TDM 1
#  endif
#endif
#ifndef HAVE_TDM
#  define HAVE_TDM 0
#endif

#if HAVE_TDM
typedef __attribute__((ext_vector_type(4))) unsigned int u32x4;
typedef __attribute__((ext_vector_type(8))) int          i32x8;
typedef __attribute__((ext_vector_type(4))) int          i32x4;

__device__ __forceinline__ void tdm_load_2d_f16(void* lds_dst, const void* gsrc,
                                                unsigned tile_w, unsigned tile_h,
                                                unsigned stride_elems)
{
    unsigned long long ga = (unsigned long long)(uintptr_t)gsrc;
    unsigned lds_off = (unsigned)(uintptr_t)lds_dst;  // generic addr low 32 = LDS offset
    unsigned td0 = stride_elems;                      // tensor width: no row clipping
    unsigned td1 = 0x100000u;                         // tensor height: no clipping
    // D# group 0 (ISA 8.3): count=1 | lds_addr | global_addr[56:0] | type=2
    u32x4 g0;
    g0[0] = 1u;
    g0[1] = lds_off;
    g0[2] = (unsigned)(ga & 0xFFFFFFFFu);
    g0[3] = (unsigned)((ga >> 32) & 0x01FFFFFFu) | (2u << 30);
    // D# group 1 (ISA 8.4): data_size=1(2B), tensor/tile dims, dim0 stride
    i32x8 g1;
    g1[0] = (int)(1u << 16);                                        // data_size=2B
    g1[1] = (int)((td0 & 0xFFFFu) << 16);                           // tensor_dim0 lo
    g1[2] = (int)(((td0 >> 16) & 0xFFFFu) | ((td1 & 0xFFFFu) << 16));
    g1[3] = (int)(((td1 >> 16) & 0xFFFFu) | ((tile_w & 0xFFFFu) << 16)); // tile_dim0
    g1[4] = (int)(tile_h & 0xFFFFu);                                // tile_dim1
    g1[5] = (int)stride_elems;                                      // dim0 stride lo
    g1[6] = 0; g1[7] = 0;
    i32x4 z4 = {0, 0, 0, 0};
#if __clang_major__ >= 23
    i32x8 z8 = {0, 0, 0, 0, 0, 0, 0, 0};
    __builtin_amdgcn_tensor_load_to_lds(g0, g1, z4, z4, z8, 0);
#else
    __builtin_amdgcn_tensor_load_to_lds(g0, g1, z4, z4, 0);
#endif
}
#endif // HAVE_TDM

// ---------------------------------------------------------------------------
// Generic WMMA GEMM: C[M,N] = A[M,K] * B[K,N] (+bias), A f32 or f16, C f16/f32.
// Output rows remapped: rr = (r/chunk_rows)*chunk_stride + r%chunk_rows + base
// ---------------------------------------------------------------------------
template <bool A_F16, bool OUT_F32>
__global__ __launch_bounds__(256)
void gemm_wmma_kernel(const void* __restrict__ Aptr, const float* __restrict__ B,
                      const float* __restrict__ bias, void* __restrict__ Cptr,
                      int M, int N, int K, int lda, int ldb, int ldc,
                      int chunk_rows, int chunk_stride, int row_base)
{
    __shared__ __align__(16) _Float16 As[64 * 32];   // [m][k]
    __shared__ __align__(16) _Float16 Bt[64 * 32];   // [n][k] (transposed)

    const int tid  = threadIdx.x;
    const int lane = tid & 31;
    const int wave = tid >> 5;
    const int m0 = blockIdx.x * 64, n0 = blockIdx.y * 64;
    const int wm  = wave >> 1;
    const int wn0 = (wave & 1) * 2;
    const int row16 = lane & 15, hi = lane >> 4;

    v8f acc0 = {}; v8f acc1 = {};

    for (int k0 = 0; k0 < K; k0 += 32) {
        if (A_F16) {
            const _Float16* A = (const _Float16*)Aptr;
            int r = tid >> 2, c = (tid & 3) * 8;
            *(v8h*)&As[r * 32 + c] =
                *(const v8h*)&A[(size_t)(m0 + r) * lda + k0 + c];
        } else {
            const float* A = (const float*)Aptr;
#pragma unroll
            for (int i = 0; i < 2; ++i) {
                int id = tid * 2 + i;
                int r = id >> 3, c = (id & 7) * 4;
                v4f vv = *(const v4f*)&A[(size_t)(m0 + r) * lda + k0 + c];
#pragma unroll
                for (int e = 0; e < 4; ++e) As[r * 32 + c + e] = (_Float16)vv[e];
            }
        }
#pragma unroll
        for (int i = 0; i < 2; ++i) {
            int id = tid * 2 + i;
            int kr = id >> 4, cb = (id & 15) * 4;
            v4f vv = *(const v4f*)&B[(size_t)(k0 + kr) * ldb + n0 + cb];
#pragma unroll
            for (int e = 0; e < 4; ++e) Bt[(cb + e) * 32 + kr] = (_Float16)vv[e];
        }
        __syncthreads();

        AFrag a, b0, b1;
        const _Float16* arow = &As[(wm * 16 + row16) * 32];
        a.h[0] = *(const v8h*)&arow[hi * 8];
        a.h[1] = *(const v8h*)&arow[16 + hi * 8];
        const _Float16* p0 = &Bt[((wn0 + 0) * 16 + row16) * 32];
        b0.h[0] = *(const v8h*)&p0[hi * 8];
        b0.h[1] = *(const v8h*)&p0[16 + hi * 8];
        const _Float16* p1 = &Bt[((wn0 + 1) * 16 + row16) * 32];
        b1.h[0] = *(const v8h*)&p1[hi * 8];
        b1.h[1] = *(const v8h*)&p1[16 + hi * 8];

        acc0 = wmma_f16(a, b0, acc0);
        acc1 = wmma_f16(a, b1, acc1);
        __syncthreads();
    }

#pragma unroll
    for (int t = 0; t < 2; ++t) {
        v8f acc = t ? acc1 : acc0;
        int cg = n0 + (wn0 + t) * 16 + row16;
#pragma unroll
        for (int v = 0; v < 8; ++v) {
            int r  = m0 + wm * 16 + v + hi * 8;
            int rr = (r / chunk_rows) * chunk_stride + (r % chunk_rows) + row_base;
            if (OUT_F32) {
                float* C = (float*)Cptr;
                C[(size_t)rr * ldc + cg] = acc[v] + (bias ? bias[cg] : 0.f);
            } else {
                _Float16* C = (_Float16*)Cptr;
                C[(size_t)rr * ldc + cg] = (_Float16)acc[v];
            }
        }
    }
}

// ---------------------------------------------------------------------------
// Fused talking-heads attention.
// Grid: (NQ/QT, b). Block: 384 threads = 12 waves, one wave per head.
// QT=32: lane == query row, softmax reductions are lane-local.
// K tiles: software-pipelined TDM (issue t+1 right after S releases the
// buffer; wave0 waits TENSORcnt at loop top). V tiles: global loads issued
// before the S WMMAs, scattered transposed into their own LDS buffer.
// Q staged in LDS (keeping it in VGPRs spilled to scratch: 12 resident waves
// cap the per-wave VGPR budget).
// ---------------------------------------------------------------------------
__global__ __launch_bounds__(384)
void attn_talking_heads_kernel(const _Float16* __restrict__ q,
                               const _Float16* __restrict__ kv,
                               const float* __restrict__ mix_pre,
                               const float* __restrict__ mix_post,
                               _Float16* __restrict__ ao)
{
    __shared__ __align__(16) _Float16 Qs[QT * DM];          // 48 KB Q tile
    __shared__ __align__(16) _Float16 KV0[JT * DM];         // 48 KB K tile
    __shared__ __align__(16) _Float16 KV1[HEADS * DH * JT]; // 48 KB Vt tile
    __shared__ __align__(16) _Float16 Sb[HEADS * QT * JT];  // 24 KB score tile (f16)
    __shared__ __align__(16) _Float16 Pb[HEADS * QT * JT];  // 24 KB post-mix attn
    __shared__ float mpre[HEADS * HEADS], mpost[HEADS * HEADS];

    const int tid  = threadIdx.x;
    const int lane = tid & 31;
    const int wave = tid >> 5;            // head index (roles h, g, g')
    const int bb = blockIdx.y, qt = blockIdx.x;
    const int row16 = lane & 15, hi = lane >> 4;

    if (tid < HEADS * HEADS) { mpre[tid] = mix_pre[tid]; mpost[tid] = mix_post[tid]; }
    __syncthreads();

    const _Float16* qsrc = &q[((size_t)bb * NQ + (size_t)qt * QT) * DM];
    const size_t kvbase = (size_t)bb * NKV * (2 * DM);
    float mold = -3.0e38f, lsum = 0.f;

#if HAVE_TDM
    if (wave == 0) {   // prime the pipeline: Q tile (contiguous) + K tile 0
        tdm_load_2d_f16(Qs, qsrc, QT * DM, 1, QT * DM);
        tdm_load_2d_f16(KV0, &kv[kvbase], DM, JT, 2 * DM);
    }
#else
    {
        const v8h* src = (const v8h*)qsrc;
        v8h* dst = (v8h*)Qs;
#pragma unroll
        for (int i = 0; i < 8; ++i) dst[tid + i * 384] = src[tid + i * 384];
    }
#endif

    // =================== pass 1: softmax statistics ===================
    for (int j0 = 0; j0 < NKV; j0 += JT) {
#if HAVE_TDM
        if (wave == 0) __builtin_amdgcn_s_wait_tensorcnt(0);
        __syncthreads();                       // Q+K(t) visible; Sb(t-1) reads done
#else
        __syncthreads();
#pragma unroll
        for (int i = 0; i < 8; ++i) {
            int id = tid + i * 384;
            int r = id / 96, cc = (id % 96) * 8;
            *(v8h*)&KV0[r * DM + cc] =
                *(const v8h*)&kv[kvbase + (size_t)(j0 + r) * (2 * DM) + cc];
        }
        __syncthreads();
#endif

        {   // S_h = (Q_h K_h^T) * scale, 32x32 per head h = wave (8 WMMAs)
            v8f c00 = {}, c01 = {}, c10 = {}, c11 = {};
#pragma unroll
            for (int k2 = 0; k2 < 2; ++k2) {
                AFrag a0, a1, b;
                const _Float16* qr0 = &Qs[row16 * DM + wave * DH + k2 * 32];
                a0.h[0] = *(const v8h*)&qr0[hi * 8];
                a0.h[1] = *(const v8h*)&qr0[16 + hi * 8];
                const _Float16* qr1 = &Qs[(16 + row16) * DM + wave * DH + k2 * 32];
                a1.h[0] = *(const v8h*)&qr1[hi * 8];
                a1.h[1] = *(const v8h*)&qr1[16 + hi * 8];
                const _Float16* pk0 = &KV0[row16 * DM + wave * DH + k2 * 32];
                b.h[0] = *(const v8h*)&pk0[hi * 8];
                b.h[1] = *(const v8h*)&pk0[16 + hi * 8];
                c00 = wmma_f16(a0, b, c00);
                c10 = wmma_f16(a1, b, c10);
                const _Float16* pk1 = &KV0[(16 + row16) * DM + wave * DH + k2 * 32];
                b.h[0] = *(const v8h*)&pk1[hi * 8];
                b.h[1] = *(const v8h*)&pk1[16 + hi * 8];
                c01 = wmma_f16(a0, b, c01);
                c11 = wmma_f16(a1, b, c11);
            }
#pragma unroll
            for (int v = 0; v < 8; ++v) {
                int m = v + hi * 8;
                Sb[(wave * QT + m) * JT + row16]           = (_Float16)(c00[v] * SCALE);
                Sb[(wave * QT + m) * JT + 16 + row16]      = (_Float16)(c01[v] * SCALE);
                Sb[(wave * QT + 16 + m) * JT + row16]      = (_Float16)(c10[v] * SCALE);
                Sb[(wave * QT + 16 + m) * JT + 16 + row16] = (_Float16)(c11[v] * SCALE);
            }
        }
        __syncthreads();                       // S done: KV0 dead, Sb ready
#if HAVE_TDM
        if (wave == 0 && j0 + JT < NKV)        // overlap DMA with mixing below
            tdm_load_2d_f16(KV0, &kv[kvbase + (size_t)(j0 + JT) * (2 * DM)],
                            DM, JT, 2 * DM);
#endif

        {   // pre-mix + online softmax stats, lane owns query row = lane
            float Dv[32];
#pragma unroll
            for (int c = 0; c < 32; ++c) Dv[c] = 0.f;
#pragma unroll
            for (int h2 = 0; h2 < HEADS; ++h2) {
                float w = mpre[h2 * HEADS + wave];
                const v8h* pr = (const v8h*)&Sb[(h2 * QT + lane) * JT];
#pragma unroll
                for (int q8 = 0; q8 < 4; ++q8) {
                    v8h hv = pr[q8];
#pragma unroll
                    for (int e = 0; e < 8; ++e) Dv[q8 * 8 + e] += w * (float)hv[e];
                }
            }
            float tmax = Dv[0];
#pragma unroll
            for (int c = 1; c < 32; ++c) tmax = fmaxf(tmax, Dv[c]);
            float mnew = fmaxf(mold, tmax);
            float se = 0.f;
#pragma unroll
            for (int c = 0; c < 32; ++c) se += __expf(Dv[c] - mnew);
            lsum = lsum * __expf(mold - mnew) + se;
            mold = mnew;
        }
    }
    const float mfin = mold;
    const float linv = 1.f / lsum;

#if HAVE_TDM
    if (wave == 0)   // prime pass 2 pipeline (KV0 is dead here)
        tdm_load_2d_f16(KV0, &kv[kvbase], DM, JT, 2 * DM);
#endif

    // =================== pass 2: attention output ===================
    v8f o0[4] = {}, o1[4] = {};
    for (int j0 = 0; j0 < NKV; j0 += JT) {
#if HAVE_TDM
        if (wave == 0) __builtin_amdgcn_s_wait_tensorcnt(0);
        __syncthreads();                       // K(t) ready; KV1/Pb(t-1) released
#else
        __syncthreads();
#pragma unroll
        for (int i = 0; i < 8; ++i) {
            int id = tid + i * 384;
            int r = id / 96, cc = (id % 96) * 8;
            *(v8h*)&KV0[r * DM + cc] =
                *(const v8h*)&kv[kvbase + (size_t)(j0 + r) * (2 * DM) + cc];
        }
        __syncthreads();
#endif

        // ---- issue V-tile global loads now; latency hides under S WMMAs
        v8h vreg[8];
        int vr[8], vc[8];
#pragma unroll
        for (int i = 0; i < 8; ++i) {
            int id = tid + i * 384;
            vr[i] = id / 96; vc[i] = (id % 96) * 8;
            vreg[i] = *(const v8h*)&kv[kvbase + (size_t)(j0 + vr[i]) * (2 * DM)
                                       + DM + vc[i]];
        }
        if (j0 + JT < NKV)                     // warm GL2 for next V tile
            __builtin_prefetch(&kv[kvbase + (size_t)(j0 + JT + (tid >> 4)) * (2 * DM)
                                   + DM + (size_t)(tid & 15) * 48], 0, 3);

        {   // recompute S_h
            v8f c00 = {}, c01 = {}, c10 = {}, c11 = {};
#pragma unroll
            for (int k2 = 0; k2 < 2; ++k2) {
                AFrag a0, a1, b;
                const _Float16* qr0 = &Qs[row16 * DM + wave * DH + k2 * 32];
                a0.h[0] = *(const v8h*)&qr0[hi * 8];
                a0.h[1] = *(const v8h*)&qr0[16 + hi * 8];
                const _Float16* qr1 = &Qs[(16 + row16) * DM + wave * DH + k2 * 32];
                a1.h[0] = *(const v8h*)&qr1[hi * 8];
                a1.h[1] = *(const v8h*)&qr1[16 + hi * 8];
                const _Float16* pk0 = &KV0[row16 * DM + wave * DH + k2 * 32];
                b.h[0] = *(const v8h*)&pk0[hi * 8];
                b.h[1] = *(const v8h*)&pk0[16 + hi * 8];
                c00 = wmma_f16(a0, b, c00);
                c10 = wmma_f16(a1, b, c10);
                const _Float16* pk1 = &KV0[(16 + row16) * DM + wave * DH + k2 * 32];
                b.h[0] = *(const v8h*)&pk1[hi * 8];
                b.h[1] = *(const v8h*)&pk1[16 + hi * 8];
                c01 = wmma_f16(a0, b, c01);
                c11 = wmma_f16(a1, b, c11);
            }
#pragma unroll
            for (int v = 0; v < 8; ++v) {
                int m = v + hi * 8;
                Sb[(wave * QT + m) * JT + row16]           = (_Float16)(c00[v] * SCALE);
                Sb[(wave * QT + m) * JT + 16 + row16]      = (_Float16)(c01[v] * SCALE);
                Sb[(wave * QT + 16 + m) * JT + row16]      = (_Float16)(c10[v] * SCALE);
                Sb[(wave * QT + 16 + m) * JT + 16 + row16] = (_Float16)(c11[v] * SCALE);
            }
        }
        __syncthreads();                       // S done: KV0 dead, Sb ready
#if HAVE_TDM
        if (wave == 0 && j0 + JT < NKV)        // overlap DMA with everything below
            tdm_load_2d_f16(KV0, &kv[kvbase + (size_t)(j0 + JT) * (2 * DM)],
                            DM, JT, 2 * DM);
#endif

        // ---- scatter V registers into transposed LDS tile: Vt[h][d][j]
#pragma unroll
        for (int i = 0; i < 8; ++i) {
#pragma unroll
            for (int e = 0; e < 8; ++e) {
                int col = vc[i] + e;
                KV1[((col >> 6) * DH + (col & 63)) * JT + vr[i]] = vreg[i][e];
            }
        }

        // ---- pre-mix + normalized softmax values
        float Av[32];
#pragma unroll
        for (int c = 0; c < 32; ++c) Av[c] = 0.f;
#pragma unroll
        for (int h2 = 0; h2 < HEADS; ++h2) {
            float w = mpre[h2 * HEADS + wave];
            const v8h* pr = (const v8h*)&Sb[(h2 * QT + lane) * JT];
#pragma unroll
            for (int q8 = 0; q8 < 4; ++q8) {
                v8h hv = pr[q8];
#pragma unroll
                for (int e = 0; e < 8; ++e) Av[q8 * 8 + e] += w * (float)hv[e];
            }
        }
#pragma unroll
        for (int c = 0; c < 32; ++c) Av[c] = __expf(Av[c] - mfin) * linv;

        __syncthreads();                       // raw-Sb reads + V scatter done
#pragma unroll
        for (int c = 0; c < 32; ++c)           // overwrite Sb with normalized A_g
            Sb[(wave * QT + lane) * JT + c] = (_Float16)Av[c];
        __syncthreads();

        {   // post-mix: M_g' = sum_g mix_post[g,g'] A_g -> Pb
            float Mv[32];
#pragma unroll
            for (int c = 0; c < 32; ++c) Mv[c] = 0.f;
#pragma unroll
            for (int g2 = 0; g2 < HEADS; ++g2) {
                float w = mpost[g2 * HEADS + wave];
                const v8h* pr = (const v8h*)&Sb[(g2 * QT + lane) * JT];
#pragma unroll
                for (int q8 = 0; q8 < 4; ++q8) {
                    v8h hv = pr[q8];
#pragma unroll
                    for (int e = 0; e < 8; ++e) Mv[q8 * 8 + e] += w * (float)hv[e];
                }
            }
#pragma unroll
            for (int c = 0; c < 32; ++c)
                Pb[(wave * QT + lane) * JT + c] = (_Float16)Mv[c];
        }
        __syncthreads();                       // Pb ready

        {   // O_g' += M_g'(32x32) x V_g'(32x64): 8 WMMAs per tile
            AFrag a0, a1, b;
            const _Float16* ar0 = &Pb[(wave * QT + row16) * JT];
            a0.h[0] = *(const v8h*)&ar0[hi * 8];
            a0.h[1] = *(const v8h*)&ar0[16 + hi * 8];
            const _Float16* ar1 = &Pb[(wave * QT + 16 + row16) * JT];
            a1.h[0] = *(const v8h*)&ar1[hi * 8];
            a1.h[1] = *(const v8h*)&ar1[16 + hi * 8];
#pragma unroll
            for (int nn = 0; nn < 4; ++nn) {
                const _Float16* brow = &KV1[(wave * DH + nn * 16 + row16) * JT];
                b.h[0] = *(const v8h*)&brow[hi * 8];
                b.h[1] = *(const v8h*)&brow[16 + hi * 8];
                o0[nn] = wmma_f16(a0, b, o0[nn]);
                o1[nn] = wmma_f16(a1, b, o1[nn]);
            }
        }
    }

    // store attention output tile as f16 for the final projection GEMM
    const size_t rowbase = (size_t)bb * NQ + (size_t)qt * QT;
#pragma unroll
    for (int nn = 0; nn < 4; ++nn) {
#pragma unroll
        for (int v = 0; v < 8; ++v) {
            int m   = v + hi * 8;
            int col = wave * DH + nn * 16 + row16;
            ao[(rowbase + m) * DM + col]      = (_Float16)o0[nn][v];
            ao[(rowbase + 16 + m) * DM + col] = (_Float16)o1[nn][v];
        }
    }
}

// ---------------------------------------------------------------------------
extern "C" void kernel_launch(void* const* d_in, const int* in_sizes, int n_in,
                              void* d_out, int out_size, void* d_ws, size_t ws_size,
                              hipStream_t stream)
{
    (void)in_sizes; (void)n_in; (void)out_size; (void)ws_size;

    const float* x     = (const float*)d_in[0];
    const float* ctx   = (const float*)d_in[1];
    const float* Wq    = (const float*)d_in[2];
    const float* Wkv   = (const float*)d_in[3];
    const float* mpre  = (const float*)d_in[4];
    const float* mpost = (const float*)d_in[5];
    const float* Wout  = (const float*)d_in[6];
    const float* bout  = (const float*)d_in[7];

    // workspace: q (f16) | kv (f16) | attn_out (f16)  ~ 38 MB
    _Float16* qf  = (_Float16*)d_ws;
    _Float16* kvf = qf  + (size_t)4 * NQ * DM;
    _Float16* aof = kvf + (size_t)4 * NKV * (2 * DM);

    dim3 blk(256);

    // q = x @ Wq                      [4096,768] x [768,768]
    gemm_wmma_kernel<false, false><<<dim3(64, 12), blk, 0, stream>>>(
        x, Wq, nullptr, qf, 4096, 768, 768, 768, 768, 768, 1024, 1024, 0);

    // kv rows from x (j = 0..1023 of each batch)
    gemm_wmma_kernel<false, false><<<dim3(64, 24), blk, 0, stream>>>(
        x, Wkv, nullptr, kvf, 4096, 1536, 768, 768, 1536, 1536, 1024, 2048, 0);

    // kv rows from context (j = 1024..2047 of each batch)
    gemm_wmma_kernel<false, false><<<dim3(64, 24), blk, 0, stream>>>(
        ctx, Wkv, nullptr, kvf, 4096, 1536, 768, 768, 1536, 1536, 1024, 2048, 1024);

    // fused talking-heads attention
    attn_talking_heads_kernel<<<dim3(NQ / QT, 4), dim3(384), 0, stream>>>(
        qf, kvf, mpre, mpost, aof);

    // out = attn_out @ Wout + b_out   (f32 output)
    gemm_wmma_kernel<true, true><<<dim3(64, 12), blk, 0, stream>>>(
        aof, Wout, bout, d_out, 4096, 768, 768, 768, 768, 768, 1024, 1024, 0);
}
